// SelfAttentionModule_62672162784048
// MI455X (gfx1250) — matrix operand order repaired and checked by reading, not compiled, not run on hardware
//
#include <hip/hip_runtime.h>
#include <hip/hip_bf16.h>
#include <math.h>

typedef __bf16 bf16;
typedef __attribute__((ext_vector_type(8)))  __bf16 bf16x8;
typedef __attribute__((ext_vector_type(16))) __bf16 v16bf;
typedef __attribute__((ext_vector_type(8)))  float  v8f;

#define BATCH 8
#define CH    256
#define NPIX  4096
#define CQK   32
#define BN_EPS 1e-5f
#define LOG2E 1.4426950408889634f

__device__ __forceinline__ v16bf mk16(bf16x8 lo, bf16x8 hi) {
    v16bf r;
#pragma unroll
    for (int e = 0; e < 8; e++) { r[e] = lo[e]; r[e + 8] = hi[e]; }
    return r;
}

// ---------------------------------------------------------------------------
// Kernel 0: fp32 -> bf16 weight conversion (one-shot, tiny)
// ---------------------------------------------------------------------------
__global__ void __launch_bounds__(256)
cvt_kernel(const float* __restrict__ src, bf16* __restrict__ dst, int n)
{
    int i = blockIdx.x * 256 + threadIdx.x;
    if (i < n) dst[i] = (bf16)src[i];
}

// ---------------------------------------------------------------------------
// Kernel 1: projections. Outputs:
//   Q [b][n][32]  pixel-major bf16, PRE-SCALED by log2(e) for exp2-domain softmax
//   Kt[b][n][32]  pixel-major bf16   (B-fragment friendly for scores)
//   Vc[b][c][n]   CHANNEL-major bf16 (B-fragment friendly for PV: contiguous j)
// ---------------------------------------------------------------------------
__global__ void __launch_bounds__(256)
proj_kernel(const float* __restrict__ x,
            const bf16* __restrict__ wqb, const float* __restrict__ bq,
            const bf16* __restrict__ wkb, const float* __restrict__ bk,
            const bf16* __restrict__ wvb, const float* __restrict__ bv,
            bf16* __restrict__ Q, bf16* __restrict__ Kt, bf16* __restrict__ Vc)
{
    __shared__ __align__(16) bf16 xT[64][CH + 8];   // 64 pixels x 256 ch (padded)

    const int t    = threadIdx.x;
    const int b    = blockIdx.y;
    const int n0   = blockIdx.x * 64;
    const int wave = t >> 5;
    const int lane = t & 31;
    const int lm   = lane & 15;
    const int lh   = lane >> 4;

    // cooperative transpose-load of x tile (coalesced over n)
    {
        const int nloc  = t & 63;
        const int cbase = t >> 6;            // 0..3
#pragma unroll 4
        for (int k = 0; k < 64; k++) {
            const int c = k * 4 + cbase;
            float v = x[((size_t)b * CH + c) * NPIX + n0 + nloc];
            xT[nloc][c] = (bf16)v;
        }
    }
    __syncthreads();

    // 80 output-tile jobs: 4 pixel tiles x (2 Q + 2 K + 16 V) 16-wide out blocks
    for (int i = 0; i < 10; i++) {
        const int job = wave * 10 + i;
        const int pt  = job / 20;            // pixel tile 0..3
        const int ot  = job % 20;

        const bf16*  W;
        const float* bias;
        int out0;
        int kind;                            // 0=Q 1=K 2=V
        if (ot < 2)       { W = wqb; bias = bq; out0 = ot * 16;        kind = 0; }
        else if (ot < 4)  { W = wkb; bias = bk; out0 = (ot - 2) * 16;  kind = 1; }
        else              { W = wvb; bias = bv; out0 = (ot - 4) * 16;  kind = 2; }

        v8f acc;
        const float bval = bias[out0 + lm];
#pragma unroll
        for (int r = 0; r < 8; r++) acc[r] = bval;

        const bf16* wrow = W + (size_t)(out0 + lm) * CH;
#pragma unroll
        for (int cb = 0; cb < 8; cb++) {
            const int ci0 = cb * 32;
            // A fragment: 16 pixels x 32 channels (k = 16*half + e)
            const bf16* ar = &xT[pt * 16 + lm][ci0];
            v16bf af = mk16(*(const bf16x8*)(ar + 16 * lh),
                            *(const bf16x8*)(ar + 16 * lh + 8));
            // B fragment: W^T block 32ci x 16out (k = 16*half + e, n = lane%16)
            const bf16* wr = wrow + ci0 + 16 * lh;
            v16bf bf_ = mk16(*(const bf16x8*)(wr),
                             *(const bf16x8*)(wr + 8));
            acc = __builtin_amdgcn_wmma_f32_16x16x32_bf16(
                false, af, false, bf_, (short)0, acc, false, false);
        }

        // D layout: m = r + 8*half (pixel), n = lane%16 (out channel)
        const float qscale = (kind == 0) ? LOG2E : 1.0f;
#pragma unroll
        for (int r = 0; r < 8; r++) {
            const int np = n0 + pt * 16 + r + 8 * lh;
            bf16 v = (bf16)(acc[r] * qscale);
            if (kind == 2)
                Vc[((size_t)b * CH + out0 + lm) * NPIX + np] = v;   // channel-major
            else if (kind == 0)
                Q [((size_t)b * NPIX + np) * CQK + out0 + lm] = v;
            else
                Kt[((size_t)b * NPIX + np) * CQK + out0 + lm] = v;
        }
    }
}

// ---------------------------------------------------------------------------
// Kernel 2: flash attention + gamma*O + x -> Opre (B,C,N) fp32
// Block = 128 query rows (8 waves x 16). 64 keys per online-softmax update:
// 4 score WMMAs + 32 PV WMMAs amortize one rescale + two shuffle reductions.
// V read straight from L2 (channel-major, contiguous 32B row segments).
// ---------------------------------------------------------------------------
__global__ void __launch_bounds__(256)
attn_kernel(const bf16* __restrict__ Q, const bf16* __restrict__ Kt,
            const bf16* __restrict__ Vc, const float* __restrict__ x,
            const float* __restrict__ gamma, float* __restrict__ Opre)
{
    __shared__ __align__(16) bf16 Plds[8][16][64];    // per-wave P scratch (16 KB)

    const int t    = threadIdx.x;
    const int b    = blockIdx.y;
    const int n0   = blockIdx.x * 128;
    const int wave = t >> 5;
    const int lane = t & 31;
    const int lm   = lane & 15;
    const int lh   = lane >> 4;
    const int qm0  = n0 + wave * 16;

    // Q A-fragment (16 rows x 32 k), rows are contiguous 64B in memory
    const bf16* qr = Q + ((size_t)b * NPIX + qm0 + lm) * CQK;
    v16bf qf = mk16(*(const bf16x8*)(qr + 16 * lh),
                    *(const bf16x8*)(qr + 16 * lh + 8));

    // per-lane base pointers for K / V fragment rows
    const bf16* kbase = Kt + ((size_t)b * NPIX + lm) * CQK + 16 * lh;
    const bf16* vbase = Vc + ((size_t)b * CH + lm) * NPIX + 16 * lh;

    v8f vzero;
#pragma unroll
    for (int r = 0; r < 8; r++) vzero[r] = 0.f;

    v8f Og[16];
#pragma unroll
    for (int g = 0; g < 16; g++)
#pragma unroll
        for (int r = 0; r < 8; r++) Og[g][r] = 0.f;
    float mrow[8], lrow[8];
#pragma unroll
    for (int r = 0; r < 8; r++) { mrow[r] = -1e30f; lrow[r] = 0.f; }

    for (int jb = 0; jb < NPIX / 64; jb++) {
        const int j0 = jb * 64;

        // prefetch next iteration's K rows into cache
        if (jb + 1 < NPIX / 64)
            __builtin_prefetch(kbase + (size_t)(j0 + 64) * CQK, 0, 3);

        // scores: four 16x16 tiles (key quarters), K=32; exp2 domain
        v8f s[4];
#pragma unroll
        for (int h = 0; h < 4; h++) {
            const bf16* kr = kbase + (size_t)(j0 + 16 * h) * CQK;
            v16bf kf = mk16(*(const bf16x8*)(kr), *(const bf16x8*)(kr + 8));
            s[h] = __builtin_amdgcn_wmma_f32_16x16x32_bf16(
                false, qf, false, kf, (short)0, vzero, false, false);
        }

        // online softmax: row m = r + 8*lh lives across the 16 lanes of this half
        float mb[8];
#pragma unroll
        for (int r = 0; r < 8; r++)
            mb[r] = fmaxf(fmaxf(s[0][r], s[1][r]), fmaxf(s[2][r], s[3][r]));
#pragma unroll
        for (int off = 1; off < 16; off <<= 1)
#pragma unroll
            for (int r = 0; r < 8; r++) mb[r] = fmaxf(mb[r], __shfl_xor(mb[r], off, 32));

        float alpha[8], sb[8];
#pragma unroll
        for (int r = 0; r < 8; r++) {
            float mn = fmaxf(mrow[r], mb[r]);
            alpha[r] = exp2f(mrow[r] - mn);
            mrow[r]  = mn;
            sb[r]    = 0.f;
        }

        // P = exp2(s - m): convert to bf16 and park in wave-private LDS tile
        // (D-layout -> A-layout transpose), accumulate row sums on the way
#pragma unroll
        for (int h = 0; h < 4; h++)
#pragma unroll
            for (int r = 0; r < 8; r++) {
                float p = exp2f(s[h][r] - mrow[r]);
                sb[r] += p;
                Plds[wave][r + 8 * lh][16 * h + lm] = (bf16)p;
            }
#pragma unroll
        for (int off = 1; off < 16; off <<= 1)
#pragma unroll
            for (int r = 0; r < 8; r++) sb[r] += __shfl_xor(sb[r], off, 32);
#pragma unroll
        for (int r = 0; r < 8; r++) lrow[r] = lrow[r] * alpha[r] + sb[r];
#pragma unroll
        for (int g = 0; g < 16; g++)
#pragma unroll
            for (int r = 0; r < 8; r++) Og[g][r] *= alpha[r];

        // two P A-fragments (j halves of the 64-key block)
        const bf16* pr = &Plds[wave][lm][0];
        v16bf pf0 = mk16(*(const bf16x8*)(pr + 16 * lh),
                         *(const bf16x8*)(pr + 16 * lh + 8));
        v16bf pf1 = mk16(*(const bf16x8*)(pr + 32 + 16 * lh),
                         *(const bf16x8*)(pr + 40 + 16 * lh));

        // O += P (16x64) * V block (64 x 256): B-fragments are contiguous rows of Vc
#pragma unroll
        for (int g = 0; g < 16; g++) {
            const bf16* vr = vbase + (size_t)(g * 16) * NPIX + j0;
            v16bf vf0 = mk16(*(const bf16x8*)(vr), *(const bf16x8*)(vr + 8));
            Og[g] = __builtin_amdgcn_wmma_f32_16x16x32_bf16(
                false, pf0, false, vf0, (short)0, Og[g], false, false);
            v16bf vf1 = mk16(*(const bf16x8*)(vr + 32), *(const bf16x8*)(vr + 40));
            Og[g] = __builtin_amdgcn_wmma_f32_16x16x32_bf16(
                false, pf1, false, vf1, (short)0, Og[g], false, false);
        }
    }

    // epilogue: normalize by l, fuse residual gamma*O + x, write pre-BN (C-major)
    const float g0 = gamma[0];
#pragma unroll
    for (int g = 0; g < 16; g++) {
        const int c = g * 16 + lm;
#pragma unroll
        for (int r = 0; r < 8; r++) {
            const int np = qm0 + r + 8 * lh;
            const size_t idx = ((size_t)b * CH + c) * NPIX + np;
            float o = Og[g][r] / lrow[r];
            Opre[idx] = g0 * o + x[idx];
        }
    }
}

// ---------------------------------------------------------------------------
// Kernel 3: per-channel BN statistics over (B, N)
// ---------------------------------------------------------------------------
__global__ void __launch_bounds__(256)
stats_kernel(const float* __restrict__ Opre, float* __restrict__ mean,
             float* __restrict__ rstd)
{
    __shared__ float r1[256], r2[256];
    const int c = blockIdx.x;
    const int t = threadIdx.x;
    float s1 = 0.f, s2 = 0.f;
    for (int idx = t; idx < BATCH * NPIX; idx += 256) {
        const int b = idx >> 12;
        const int n = idx & (NPIX - 1);
        float v = Opre[((size_t)b * CH + c) * NPIX + n];
        s1 += v; s2 += v * v;
    }
    r1[t] = s1; r2[t] = s2;
    __syncthreads();
    for (int off = 128; off > 0; off >>= 1) {
        if (t < off) { r1[t] += r1[t + off]; r2[t] += r2[t + off]; }
        __syncthreads();
    }
    if (t == 0) {
        const float inv = 1.0f / (float)(BATCH * NPIX);
        float m = r1[0] * inv;
        float var = r2[0] * inv - m * m;
        mean[c] = m;
        rstd[c] = rsqrtf(var + BN_EPS);
    }
}

// ---------------------------------------------------------------------------
// Kernel 4: normalize + affine + ReLU -> d_out
// ---------------------------------------------------------------------------
__global__ void __launch_bounds__(256)
finalize_kernel(const float* __restrict__ Opre, const float* __restrict__ mean,
                const float* __restrict__ rstd, const float* __restrict__ bnw,
                const float* __restrict__ bnb, float* __restrict__ out)
{
    const size_t idx = (size_t)blockIdx.x * 256 + threadIdx.x;
    const int c = (int)((idx >> 12) & (CH - 1));
    float v = Opre[idx];
    float o = (v - mean[c]) * rstd[c] * bnw[c] + bnb[c];
    out[idx] = fmaxf(o, 0.f);
}

// ---------------------------------------------------------------------------
extern "C" void kernel_launch(void* const* d_in, const int* in_sizes, int n_in,
                              void* d_out, int out_size, void* d_ws, size_t ws_size,
                              hipStream_t stream)
{
    const float* x     = (const float*)d_in[0];
    const float* wq    = (const float*)d_in[1];
    const float* bq    = (const float*)d_in[2];
    const float* wk    = (const float*)d_in[3];
    const float* bk    = (const float*)d_in[4];
    const float* wv    = (const float*)d_in[5];
    const float* bv    = (const float*)d_in[6];
    const float* gamma = (const float*)d_in[7];
    const float* bnw   = (const float*)d_in[8];
    const float* bnb   = (const float*)d_in[9];
    float* out = (float*)d_out;

    char* ws = (char*)d_ws;
    size_t offQ  = 0;
    size_t offK  = offQ  + (size_t)BATCH * NPIX * CQK * sizeof(bf16);
    size_t offV  = offK  + (size_t)BATCH * NPIX * CQK * sizeof(bf16);
    size_t offO  = offV  + (size_t)BATCH * NPIX * CH  * sizeof(bf16);
    size_t offM  = offO  + (size_t)BATCH * CH * NPIX * sizeof(float);
    size_t offR  = offM  + CH * sizeof(float);
    size_t offWQ = offR  + CH * sizeof(float);
    size_t offWK = offWQ + (size_t)CQK * CH * sizeof(bf16);
    size_t offWV = offWK + (size_t)CQK * CH * sizeof(bf16);

    bf16*  Q    = (bf16*)(ws + offQ);
    bf16*  Kt   = (bf16*)(ws + offK);
    bf16*  Vc   = (bf16*)(ws + offV);
    float* Opre = (float*)(ws + offO);
    float* mean = (float*)(ws + offM);
    float* rstd = (float*)(ws + offR);
    bf16*  wqb  = (bf16*)(ws + offWQ);
    bf16*  wkb  = (bf16*)(ws + offWK);
    bf16*  wvb  = (bf16*)(ws + offWV);

    cvt_kernel<<<(CQK * CH + 255) / 256, 256, 0, stream>>>(wq, wqb, CQK * CH);
    cvt_kernel<<<(CQK * CH + 255) / 256, 256, 0, stream>>>(wk, wkb, CQK * CH);
    cvt_kernel<<<(CH * CH + 255) / 256, 256, 0, stream>>>(wv, wvb, CH * CH);

    proj_kernel<<<dim3(NPIX / 64, BATCH), 256, 0, stream>>>(
        x, wqb, bq, wkb, bk, wvb, bv, Q, Kt, Vc);

    attn_kernel<<<dim3(NPIX / 128, BATCH), 256, 0, stream>>>(
        Q, Kt, Vc, x, gamma, Opre);

    stats_kernel<<<CH, 256, 0, stream>>>(Opre, mean, rstd);

    finalize_kernel<<<(BATCH * CH * NPIX) / 256, 256, 0, stream>>>(
        Opre, mean, rstd, bnw, bnb, out);
}
